// LSTMSeq2Seq_37417755082991
// MI455X (gfx1250) — matrix operand order repaired and checked by reading, not compiled, not run on hardware
//
#include <hip/hip_runtime.h>

typedef __bf16 v16bf __attribute__((ext_vector_type(16)));
typedef __bf16 v8bf  __attribute__((ext_vector_type(8)));
typedef float  v8f   __attribute__((ext_vector_type(8)));

#define BATCH 256
#define HID   512
#define SEQT  512
#define DIN   64
#define DOUT  64
#define FUT   96

// ---------------------------------------------------------------- helpers
static __device__ __forceinline__ unsigned short f2bf_rne(float f) {
  union { float f; unsigned u; } v; v.f = f;
  unsigned u = v.u;
  unsigned r = u + 0x7FFFu + ((u >> 16) & 1u);
  return (unsigned short)(r >> 16);
}

static __device__ __forceinline__ float sigmoidf_(float x) {
  return 1.0f / (1.0f + __expf(-x));
}

static __device__ __forceinline__ v8f wmma_bf16(v8f acc, v16bf a, v16bf b) {
  return __builtin_amdgcn_wmma_f32_16x16x32_bf16(
      false, a, false, b, (short)0, acc, false, false);
}

// Build a 16x32 A-fragment from bf16 row-major data.
// Lane l (hi=l>>4): elements 0..7 = row[8*hi+0..7], elements 8..15 = row[16+8*hi+0..7]
// (caller passes p = rowBase + kc*32 + 8*hi, in bf16-element units)
static __device__ __forceinline__ v16bf load_a_frag(const unsigned short* __restrict__ p) {
  v8bf lo = *(const v8bf*)(p);
  v8bf hh = *(const v8bf*)(p + 16);
  v16bf a;
  a[0]  = lo[0]; a[1]  = lo[1]; a[2]  = lo[2]; a[3]  = lo[3];
  a[4]  = lo[4]; a[5]  = lo[5]; a[6]  = lo[6]; a[7]  = lo[7];
  a[8]  = hh[0]; a[9]  = hh[1]; a[10] = hh[2]; a[11] = hh[3];
  a[12] = hh[4]; a[13] = hh[5]; a[14] = hh[6]; a[15] = hh[7];
  return a;
}

// ---------------------------------------------------------------- prep kernels
// Repack f32 weight W[R x C] (row-major; gates = x @ W^T) into bf16 B-fragment
// layout: out[((jtg*nk + kc)*32 + lane)*16 + e] = bf16(W[(jtg*16 + (lane&15))*C
//                                                      + kc*32 + 8*(lane>>4) + (e&7) + 16*(e>>3)])
__global__ void pack_w(const float* __restrict__ W, unsigned short* __restrict__ out,
                       int C, int nk) {
  int idx  = blockIdx.x * blockDim.x + threadIdx.x;
  int e    = idx & 15;
  int lane = (idx >> 4) & 31;
  int tile = idx >> 9;            // jtg*nk + kc
  int kc   = tile % nk;
  int jtg  = tile / nk;
  int n    = jtg * 16 + (lane & 15);
  int k    = kc * 32 + 8 * (lane >> 4) + (e & 7) + 16 * (e >> 3);
  out[idx] = f2bf_rne(W[(size_t)n * C + k]);
}

__global__ void cvt_bf16(const float* __restrict__ in, unsigned short* __restrict__ out,
                         int n) {
  int i = blockIdx.x * blockDim.x + threadIdx.x;
  if (i < n) out[i] = f2bf_rne(in[i]);
}

__global__ void vec_add(const float* __restrict__ a, const float* __restrict__ b,
                        float* __restrict__ o, int n) {
  int i = blockIdx.x * blockDim.x + threadIdx.x;
  if (i < n) o[i] = a[i] + b[i];
}

__global__ void vec_zero(float* __restrict__ o, int n) {
  int i = blockIdx.x * blockDim.x + threadIdx.x;
  if (i < n) o[i] = 0.0f;
}

// ---------------------------------------------------------------- LSTM step
// gates = x @ Wih^T + hprev @ Whh^T + bias ; then fused i,f,g,o nonlinearity.
// x/hprev/hnext are bf16 row-major; c state stays f32.
// One wave per 16x16 (m,j) output tile; computes all four gate tiles so the
// elementwise c/h update is wave-local. 64 blocks x 256 threads = 512 waves
// = 16 m-tiles x 32 j-tiles.
__global__ __launch_bounds__(256) void lstm_step(
    const unsigned short* __restrict__ x, int ldx, int nkx,
    const v16bf* __restrict__ wih, const v16bf* __restrict__ whh,
    const float* __restrict__ bias,
    const unsigned short* __restrict__ hprev, unsigned short* __restrict__ hnext,
    float* __restrict__ cst) {
  int lane = threadIdx.x & 31;
  int wave = threadIdx.x >> 5;
  int tid  = blockIdx.x * 8 + wave;   // 0..511
  int mt   = tid >> 5;                // 0..15
  int jt   = tid & 31;                // 0..31
  int hi   = lane >> 4;
  int ml   = lane & 15;

  v8f ai = {}, af = {}, ag = {}, ao = {};

  // ---- input projection: K = nkx*32 (64 or 512)
  const unsigned short* xrow = x + (size_t)(mt * 16 + ml) * ldx + 8 * hi;
  for (int kc = 0; kc < nkx; ++kc) {
    v16bf a  = load_a_frag(xrow + kc * 32);
    v16bf b0 = wih[((size_t)(0 * 32 + jt) * nkx + kc) * 32 + lane];
    v16bf b1 = wih[((size_t)(1 * 32 + jt) * nkx + kc) * 32 + lane];
    v16bf b2 = wih[((size_t)(2 * 32 + jt) * nkx + kc) * 32 + lane];
    v16bf b3 = wih[((size_t)(3 * 32 + jt) * nkx + kc) * 32 + lane];
    ai = wmma_bf16(ai, a, b0);
    af = wmma_bf16(af, a, b1);
    ag = wmma_bf16(ag, a, b2);
    ao = wmma_bf16(ao, a, b3);
  }

  // ---- recurrent projection: K = 512
  const unsigned short* hrow = hprev + (size_t)(mt * 16 + ml) * HID + 8 * hi;
  for (int kc = 0; kc < 16; ++kc) {
    v16bf a  = load_a_frag(hrow + kc * 32);
    v16bf b0 = whh[((size_t)(0 * 32 + jt) * 16 + kc) * 32 + lane];
    v16bf b1 = whh[((size_t)(1 * 32 + jt) * 16 + kc) * 32 + lane];
    v16bf b2 = whh[((size_t)(2 * 32 + jt) * 16 + kc) * 32 + lane];
    v16bf b3 = whh[((size_t)(3 * 32 + jt) * 16 + kc) * 32 + lane];
    ai = wmma_bf16(ai, a, b0);
    af = wmma_bf16(af, a, b1);
    ag = wmma_bf16(ag, a, b2);
    ao = wmma_bf16(ao, a, b3);
  }

  // ---- fused LSTM cell elementwise (f32). C/D layout: elem (M = r+8*hi, N = ml)
  int   j  = jt * 16 + ml;
  float bi = bias[j];
  float bf = bias[512 + j];
  float bg = bias[1024 + j];
  float bo = bias[1536 + j];
#pragma unroll
  for (int r = 0; r < 8; ++r) {
    int    m  = mt * 16 + r + 8 * hi;
    size_t o  = (size_t)m * HID + j;
    float  gi = ai[r] + bi;
    float  gf = af[r] + bf;
    float  gg = ag[r] + bg;
    float  go = ao[r] + bo;
    float  cn = sigmoidf_(gf) * cst[o] + sigmoidf_(gi) * tanhf(gg);
    float  hn = sigmoidf_(go) * tanhf(cn);
    cst[o]   = cn;
    hnext[o] = f2bf_rne(hn);
  }
}

// ---------------------------------------------------------------- FC head
// out = h @ fcW^T + fcb ; writes d_out[:, t, :] (f32, stride ostride) and the
// bf16 feedback buffer xnext. 8 blocks x 256 threads = 64 waves.
__global__ __launch_bounds__(256) void fc_step(
    const unsigned short* __restrict__ h, const v16bf* __restrict__ w,
    const float* __restrict__ b, float* __restrict__ out, int ostride,
    unsigned short* __restrict__ xnext) {
  int lane = threadIdx.x & 31;
  int wave = threadIdx.x >> 5;
  int tid  = blockIdx.x * 8 + wave;   // 0..63
  int mt   = tid >> 2;
  int dt   = tid & 3;
  int hi   = lane >> 4;
  int ml   = lane & 15;

  v8f acc = {};
  const unsigned short* hrow = h + (size_t)(mt * 16 + ml) * HID + 8 * hi;
  for (int kc = 0; kc < 16; ++kc) {
    v16bf a  = load_a_frag(hrow + kc * 32);
    v16bf bw = w[((size_t)dt * 16 + kc) * 32 + lane];
    acc = wmma_bf16(acc, a, bw);
  }

  int   d  = dt * 16 + ml;
  float bb = b[d];
#pragma unroll
  for (int r = 0; r < 8; ++r) {
    int   m = mt * 16 + r + 8 * hi;
    float v = acc[r] + bb;
    out[(size_t)m * ostride + d] = v;
    xnext[(size_t)m * DOUT + d]  = f2bf_rne(v);
  }
}

// ---------------------------------------------------------------- launch
extern "C" void kernel_launch(void* const* d_in, const int* in_sizes, int n_in,
                              void* d_out, int out_size, void* d_ws, size_t ws_size,
                              hipStream_t stream) {
  (void)in_sizes; (void)n_in; (void)out_size; (void)ws_size;
  const float* in_seq = (const float*)d_in[0];
  const float* eWih0  = (const float*)d_in[1];
  const float* eWhh0  = (const float*)d_in[2];
  const float* ebih0  = (const float*)d_in[3];
  const float* ebhh0  = (const float*)d_in[4];
  const float* eWih1  = (const float*)d_in[5];
  const float* eWhh1  = (const float*)d_in[6];
  const float* ebih1  = (const float*)d_in[7];
  const float* ebhh1  = (const float*)d_in[8];
  const float* dWih0  = (const float*)d_in[9];
  const float* dWhh0  = (const float*)d_in[10];
  const float* dbih0  = (const float*)d_in[11];
  const float* dbhh0  = (const float*)d_in[12];
  const float* dWih1  = (const float*)d_in[13];
  const float* dWhh1  = (const float*)d_in[14];
  const float* dbih1  = (const float*)d_in[15];
  const float* dbhh1  = (const float*)d_in[16];
  const float* fcW    = (const float*)d_in[17];
  const float* fcb    = (const float*)d_in[18];
  float* out = (float*)d_out;

  // ---- workspace carve-out (all offsets 256B aligned)
  char*  ws  = (char*)d_ws;
  size_t off = 0;
  auto carve = [&](size_t bytes) -> char* {
    char* p = ws + off;
    off = (off + bytes + 255) & ~(size_t)255;
    return p;
  };
  unsigned short* p_eWih0 = (unsigned short*)carve((size_t)2048 * 64 * 2);
  unsigned short* p_eWhh0 = (unsigned short*)carve((size_t)2048 * 512 * 2);
  unsigned short* p_eWih1 = (unsigned short*)carve((size_t)2048 * 512 * 2);
  unsigned short* p_eWhh1 = (unsigned short*)carve((size_t)2048 * 512 * 2);
  unsigned short* p_dWih0 = (unsigned short*)carve((size_t)2048 * 64 * 2);
  unsigned short* p_dWhh0 = (unsigned short*)carve((size_t)2048 * 512 * 2);
  unsigned short* p_dWih1 = (unsigned short*)carve((size_t)2048 * 512 * 2);
  unsigned short* p_dWhh1 = (unsigned short*)carve((size_t)2048 * 512 * 2);
  unsigned short* p_fc    = (unsigned short*)carve((size_t)64 * 512 * 2);
  float* b_e0 = (float*)carve(2048 * 4);
  float* b_e1 = (float*)carve(2048 * 4);
  float* b_d0 = (float*)carve(2048 * 4);
  float* b_d1 = (float*)carve(2048 * 4);
  unsigned short* h0b[2];
  unsigned short* h1b[2];
  h0b[0] = (unsigned short*)carve((size_t)BATCH * HID * 2);
  h0b[1] = (unsigned short*)carve((size_t)BATCH * HID * 2);
  h1b[0] = (unsigned short*)carve((size_t)BATCH * HID * 2);
  h1b[1] = (unsigned short*)carve((size_t)BATCH * HID * 2);
  float* c0  = (float*)carve((size_t)BATCH * HID * 4);
  float* c1  = (float*)carve((size_t)BATCH * HID * 4);
  unsigned short* xbf    = (unsigned short*)carve((size_t)BATCH * DOUT * 2);
  unsigned short* xin_bf = (unsigned short*)carve((size_t)BATCH * SEQT * DIN * 2);

  // ---- one-time (per call): pack weights, bf16 input, fuse biases, zero state
  pack_w<<<(2048 * 64) / 256, 256, 0, stream>>>(eWih0, p_eWih0, 64, 2);
  pack_w<<<(2048 * 512) / 256, 256, 0, stream>>>(eWhh0, p_eWhh0, 512, 16);
  pack_w<<<(2048 * 512) / 256, 256, 0, stream>>>(eWih1, p_eWih1, 512, 16);
  pack_w<<<(2048 * 512) / 256, 256, 0, stream>>>(eWhh1, p_eWhh1, 512, 16);
  pack_w<<<(2048 * 64) / 256, 256, 0, stream>>>(dWih0, p_dWih0, 64, 2);
  pack_w<<<(2048 * 512) / 256, 256, 0, stream>>>(dWhh0, p_dWhh0, 512, 16);
  pack_w<<<(2048 * 512) / 256, 256, 0, stream>>>(dWih1, p_dWih1, 512, 16);
  pack_w<<<(2048 * 512) / 256, 256, 0, stream>>>(dWhh1, p_dWhh1, 512, 16);
  pack_w<<<(64 * 512) / 256, 256, 0, stream>>>(fcW, p_fc, 512, 16);

  const int NIN = BATCH * SEQT * DIN;  // 8388608
  cvt_bf16<<<NIN / 256, 256, 0, stream>>>(in_seq, xin_bf, NIN);

  vec_add<<<8, 256, 0, stream>>>(ebih0, ebhh0, b_e0, 2048);
  vec_add<<<8, 256, 0, stream>>>(ebih1, ebhh1, b_e1, 2048);
  vec_add<<<8, 256, 0, stream>>>(dbih0, dbhh0, b_d0, 2048);
  vec_add<<<8, 256, 0, stream>>>(dbih1, dbhh1, b_d1, 2048);

  const int SN = BATCH * HID;  // 131072 f32 elems ; bf16 buffers = SN/2 f32-words
  vec_zero<<<(SN / 2) / 256, 256, 0, stream>>>((float*)h0b[0], SN / 2);
  vec_zero<<<(SN / 2) / 256, 256, 0, stream>>>((float*)h1b[0], SN / 2);
  vec_zero<<<SN / 256, 256, 0, stream>>>(c0, SN);
  vec_zero<<<SN / 256, 256, 0, stream>>>(c1, SN);

  int i0 = 0, i1 = 0;

  // ---- encoder: 512 steps, layers fused per-step via stream order
  for (int t = 0; t < SEQT; ++t) {
    lstm_step<<<64, 256, 0, stream>>>(xin_bf + (size_t)t * DIN, SEQT * DIN, 2,
                                      (const v16bf*)p_eWih0, (const v16bf*)p_eWhh0,
                                      b_e0, h0b[i0], h0b[1 - i0], c0);
    lstm_step<<<64, 256, 0, stream>>>(h0b[1 - i0], HID, 16,
                                      (const v16bf*)p_eWih1, (const v16bf*)p_eWhh1,
                                      b_e1, h1b[i1], h1b[1 - i1], c1);
    i0 ^= 1; i1 ^= 1;
  }

  // ---- decoder: 96 steps; step 0 reads packed input_seq[:, -1, :64] in place
  for (int t = 0; t < FUT; ++t) {
    const unsigned short* xp = (t == 0) ? (xin_bf + (size_t)(SEQT - 1) * DIN) : xbf;
    int ldx                  = (t == 0) ? SEQT * DIN : DOUT;
    lstm_step<<<64, 256, 0, stream>>>(xp, ldx, 2,
                                      (const v16bf*)p_dWih0, (const v16bf*)p_dWhh0,
                                      b_d0, h0b[i0], h0b[1 - i0], c0);
    lstm_step<<<64, 256, 0, stream>>>(h0b[1 - i0], HID, 16,
                                      (const v16bf*)p_dWih1, (const v16bf*)p_dWhh1,
                                      b_d1, h1b[i1], h1b[1 - i1], c1);
    i0 ^= 1; i1 ^= 1;
    fc_step<<<8, 256, 0, stream>>>(h1b[i1], (const v16bf*)p_fc, fcb,
                                   out + (size_t)t * DOUT, FUT * DOUT, xbf);
  }
}
